// HybridKANModel_9165460210148
// MI455X (gfx1250) — matrix operand order repaired and checked by reading, not compiled, not run on hardware
//
#include <hip/hip_runtime.h>
#include <hip/hip_bf16.h>

typedef __attribute__((ext_vector_type(16))) _Float16 v16h;
typedef __attribute__((ext_vector_type(8)))  float    v8f;

#define B_TOTAL 4096
#define KDIM    12800   // 256 features * 50 grid points
#define GRIDN   50

// ---------------------------------------------------------------------------
// Kernel 1: conv1 (3x3, 1->32) + relu + maxpool2  -> h1 (B,6,6,32)
// ---------------------------------------------------------------------------
__global__ __launch_bounds__(256) void conv1_relu_pool(
    const float* __restrict__ x,      // (B,14,14,1)
    const float* __restrict__ w,      // (3,3,1,32) HWIO
    const float* __restrict__ bias,   // (32)
    float* __restrict__ h1)           // (B,6,6,32)
{
    int idx = blockIdx.x * blockDim.x + threadIdx.x;
    if (idx >= B_TOTAL * 6 * 6 * 32) return;
    int c   = idx & 31;
    int tmp = idx >> 5;
    int pw  = tmp % 6; tmp /= 6;
    int ph  = tmp % 6; tmp /= 6;
    int b   = tmp;

    float wk[9];
#pragma unroll
    for (int t = 0; t < 9; ++t) wk[t] = w[t * 32 + c];
    float bc = bias[c];

    const float* xb = x + b * 196;   // 14*14
    float best = -3.402823466e+38f;
#pragma unroll
    for (int dy = 0; dy < 2; ++dy) {
#pragma unroll
        for (int dx = 0; dx < 2; ++dx) {
            int oh = 2 * ph + dy, ow = 2 * pw + dx;
            float acc = bc;
#pragma unroll
            for (int ky = 0; ky < 3; ++ky)
#pragma unroll
                for (int kx = 0; kx < 3; ++kx)
                    acc = __builtin_fmaf(xb[(oh + ky) * 14 + (ow + kx)],
                                         wk[ky * 3 + kx], acc);
            acc = fmaxf(acc, 0.0f);          // relu
            best = fmaxf(best, acc);         // maxpool
        }
    }
    h1[idx] = best;
}

// ---------------------------------------------------------------------------
// Kernel 2: conv2 (3x3, 32->64) + relu + maxpool2 + flatten -> h (B,256)
// One block (256 threads) per image; image staged in LDS.
// Flatten order d = (ph*2+pw)*64 + c  (keras H,W,C)
// ---------------------------------------------------------------------------
__global__ __launch_bounds__(256) void conv2_relu_pool(
    const float* __restrict__ h1,     // (B,6,6,32)
    const float* __restrict__ w,      // (3,3,32,64) HWIO
    const float* __restrict__ bias,   // (64)
    float* __restrict__ h)            // (B,256)
{
    __shared__ float img[6 * 6 * 32];
    int b   = blockIdx.x;
    int tid = threadIdx.x;
    for (int i = tid; i < 1152; i += 256) img[i] = h1[b * 1152 + i];
    __syncthreads();

    int c   = tid & 63;
    int pos = tid >> 6;        // 0..3
    int pw  = pos & 1;
    int ph  = pos >> 1;
    float bc = bias[c];

    float best = -3.402823466e+38f;
#pragma unroll
    for (int dy = 0; dy < 2; ++dy) {
#pragma unroll
        for (int dx = 0; dx < 2; ++dx) {
            int oh = 2 * ph + dy, ow = 2 * pw + dx;
            float acc = bc;
#pragma unroll
            for (int ky = 0; ky < 3; ++ky) {
#pragma unroll
                for (int kx = 0; kx < 3; ++kx) {
                    const float* ip = &img[((oh + ky) * 6 + (ow + kx)) * 32];
                    const float* wp = &w[((ky * 3 + kx) * 32) * 64 + c];
#pragma unroll
                    for (int ci = 0; ci < 32; ++ci)
                        acc = __builtin_fmaf(ip[ci], wp[ci * 64], acc);
                }
            }
            acc = fmaxf(acc, 0.0f);
            best = fmaxf(best, acc);
        }
    }
    h[b * 256 + tid] = best;
}

// ---------------------------------------------------------------------------
// Kernel 3: convert kan1_w (128,256,50) f32 -> f16, row-major [o][k]
// ---------------------------------------------------------------------------
__global__ __launch_bounds__(256) void w_to_f16(
    const float* __restrict__ src, _Float16* __restrict__ dst, int n)
{
    int i = blockIdx.x * blockDim.x + threadIdx.x;
    if (i < n) dst[i] = (_Float16)src[i];
}

// ---------------------------------------------------------------------------
// Hat-basis A-fragment builder (contiguous K order, matching the B loads):
//  element i of the per-lane v16h maps to K = kt + kHalf + i
//  a 32-wide K window crosses at most one feature boundary (grid size 50 > 32)
// ---------------------------------------------------------------------------
__device__ __forceinline__ v16h build_hat(const _Float16* hsRow,
                                          const float* gs,
                                          int kt, int kHalf)
{
    int d0 = kt / 50;
    int boundary = (d0 + 1) * 50;
    int d1 = (d0 + 1 < 256) ? d0 + 1 : 255;
    float x0 = (float)hsRow[d0];
    float x1 = (float)hsRow[d1];
    v16h a;
#pragma unroll
    for (int i = 0; i < 16; ++i) {
        int k  = kt + kHalf + i;
        bool hi = (k >= boundary);
        float x = hi ? x1 : x0;
        int   g = k - (hi ? boundary : d0 * 50);
        float t = 1.0f - __builtin_fabsf(x - gs[g]);
        a[i] = (_Float16)__builtin_fmaxf(t, 0.0f);
    }
    return a;
}

// ---------------------------------------------------------------------------
// Kernel 4: KAN layer 1 as a software-pipelined WMMA GEMM.
//   out1[b,o] = sum_k hat(h[b, k/50] - grid[k%50]) * Wh[o,k],  K = 12800
// Block: 128 threads = 4 waves (one per SIMD32 of a WGP), BM = 64 rows.
// Each wave: 16 rows x 128 cols -> 8x v8f accumulators.
// K loop unrolled by 2 with register double-buffering of the 8 B fragments
// (and the A fragment), so the 16-load clause for step t+1 is in flight
// while the 8 WMMAs of step t execute.
// ---------------------------------------------------------------------------
__global__ __launch_bounds__(128) void kan1_wmma(
    const float* __restrict__ hIn,    // (B,256)
    const float* __restrict__ grid,   // (50)
    const _Float16* __restrict__ Wh,  // (128,12800) f16 row-major
    float* __restrict__ out1)         // (B,128)
{
    __shared__ _Float16 hs[64 * 256];   // 32 KB
    __shared__ float    gs[GRIDN];

    const int tid  = threadIdx.x;
    const int lane = tid & 31;
    const int wave = tid >> 5;               // 0..3
    const int rowBase = blockIdx.x * 64;

    // stage h tile (f32 -> f16) and grid into LDS
    for (int i = tid; i < 64 * 256; i += 128)
        hs[i] = (_Float16)hIn[rowBase * 256 + i];
    if (tid < GRIDN) gs[tid] = grid[tid];
    __syncthreads();

    v8f zero = {};
    v8f c[8];
#pragma unroll
    for (int n = 0; n < 8; ++n) c[n] = zero;

    // fragment coordinates (K contiguous per lane half, same for A and B)
    const int mRow  = wave * 16 + (lane & 15);   // A: M = lane%16
    const int kHalf = (lane >> 4) * 16;          // A: lanes>=16 hold K=16..31
    const int bN    = lane & 15;                 // B: N = lane%16
    const int bK    = (lane >> 4) * 16;          // B: lanes>=16 hold K=16..31

    const _Float16* wbase = Wh + bN * KDIM + bK;
    const _Float16* hsRow = &hs[mRow * 256];

    // ---- prologue: buffer 0 holds step kt=0 ----
    v16h b0[8], b1[8];
#pragma unroll
    for (int n = 0; n < 8; ++n)
        b0[n] = *(const v16h*)(wbase + n * 16 * KDIM);
    v16h a0 = build_hat(hsRow, gs, 0, kHalf);
    v16h a1;

    for (int kt = 0; kt < KDIM; kt += 64) {
        // issue loads + VALU for step kt+32 BEFORE consuming step kt
        const _Float16* wp1 = wbase + kt + 32;
#pragma unroll
        for (int n = 0; n < 8; ++n)
            b1[n] = *(const v16h*)(wp1 + n * 16 * KDIM);
        a1 = build_hat(hsRow, gs, kt + 32, kHalf);

#pragma unroll
        for (int n = 0; n < 8; ++n)
            c[n] = __builtin_amdgcn_wmma_f32_16x16x32_f16(
                       false, a0, false, b0[n], (short)0, c[n], false, false);

        // issue loads + VALU for step kt+64 BEFORE consuming step kt+32
        if (kt + 64 < KDIM) {
            const _Float16* wp0 = wbase + kt + 64;
#pragma unroll
            for (int n = 0; n < 8; ++n)
                b0[n] = *(const v16h*)(wp0 + n * 16 * KDIM);
            a0 = build_hat(hsRow, gs, kt + 64, kHalf);
            __builtin_prefetch(wp0 + 512, 0, 0);   // ~1KB ahead, L2 streaming
        }

#pragma unroll
        for (int n = 0; n < 8; ++n)
            c[n] = __builtin_amdgcn_wmma_f32_16x16x32_f16(
                       false, a1, false, b1[n], (short)0, c[n], false, false);
    }

    // ---- epilogue: C/D layout M = r + 8*(lane>>4), N = n*16 + lane%16 ----
    const int mOff = (lane >> 4) * 8;
#pragma unroll
    for (int n = 0; n < 8; ++n) {
#pragma unroll
        for (int r = 0; r < 8; ++r) {
            int gr = rowBase + wave * 16 + r + mOff;
            out1[gr * 128 + n * 16 + (lane & 15)] = c[n][r];
        }
    }
}

// ---------------------------------------------------------------------------
// Kernel 5: relu -> SimplifiedKAN (64,128) -> dense (64,10)+bias -> softmax
// One block (128 threads) per batch row.
// ---------------------------------------------------------------------------
__global__ __launch_bounds__(128) void kan2_dense_softmax(
    const float* __restrict__ out1,   // (B,128)
    const float* __restrict__ w2,     // (64,128)
    const float* __restrict__ dw,     // (64,10)
    const float* __restrict__ db,     // (10)
    float* __restrict__ out)          // (B,10)
{
    __shared__ float s1[128];
    __shared__ float s2[64];
    __shared__ float lg[10];
    int b = blockIdx.x;
    int t = threadIdx.x;

    s1[t] = fmaxf(out1[b * 128 + t], 0.0f);
    __syncthreads();

    if (t < 64) {
        float acc = 0.0f;
        const float* wr = w2 + t * 128;
#pragma unroll 8
        for (int d = 0; d < 128; ++d) acc = __builtin_fmaf(wr[d], s1[d], acc);
        s2[t] = acc;
    }
    __syncthreads();

    if (t < 10) {
        float acc = db[t];
#pragma unroll 8
        for (int d = 0; d < 64; ++d) acc = __builtin_fmaf(s2[d], dw[d * 10 + t], acc);
        lg[t] = acc;
    }
    __syncthreads();

    if (t < 10) {
        float m = lg[0];
#pragma unroll
        for (int i = 1; i < 10; ++i) m = fmaxf(m, lg[i]);
        float s = 0.0f;
#pragma unroll
        for (int i = 0; i < 10; ++i) s += __expf(lg[i] - m);
        out[b * 10 + t] = __expf(lg[t] - m) / s;
    }
}

// ---------------------------------------------------------------------------
extern "C" void kernel_launch(void* const* d_in, const int* in_sizes, int n_in,
                              void* d_out, int out_size, void* d_ws, size_t ws_size,
                              hipStream_t stream)
{
    const float* x       = (const float*)d_in[0];   // (B,14,14,1)
    const float* conv1_w = (const float*)d_in[1];   // (3,3,1,32)
    const float* conv1_b = (const float*)d_in[2];   // (32)
    const float* conv2_w = (const float*)d_in[3];   // (3,3,32,64)
    const float* conv2_b = (const float*)d_in[4];   // (64)
    const float* gridv   = (const float*)d_in[5];   // (50)
    const float* kan1_w  = (const float*)d_in[6];   // (128,256,50)
    const float* kan2_w  = (const float*)d_in[7];   // (64,128)
    const float* dense_w = (const float*)d_in[8];   // (64,10)
    const float* dense_b = (const float*)d_in[9];   // (10)
    float* outp          = (float*)d_out;           // (B,10)

    // workspace partition (all offsets 256B-aligned)
    char* ws = (char*)d_ws;
    _Float16* Wh  = (_Float16*)(ws);                          // 3,276,800 B
    float*    h1  = (float*)(ws + 3276800);                   // 18,874,368 B
    float*    h   = (float*)(ws + 3276800 + 18874368);        //  4,194,304 B
    float*    o1  = (float*)(ws + 3276800 + 18874368 + 4194304); // 2,097,152 B

    // 1) conv1 + relu + pool
    {
        int n = B_TOTAL * 6 * 6 * 32;
        conv1_relu_pool<<<(n + 255) / 256, 256, 0, stream>>>(x, conv1_w, conv1_b, h1);
    }
    // 2) conv2 + relu + pool + flatten
    conv2_relu_pool<<<B_TOTAL, 256, 0, stream>>>(h1, conv2_w, conv2_b, h);
    // 3) kan1_w -> f16
    {
        int n = 128 * KDIM;
        w_to_f16<<<(n + 255) / 256, 256, 0, stream>>>(kan1_w, Wh, n);
    }
    // 4) fused hat-basis + software-pipelined WMMA GEMM (M=4096,N=128,K=12800)
    kan1_wmma<<<B_TOTAL / 64, 128, 0, stream>>>(h, gridv, Wh, o1);
    // 5) relu -> KAN2 -> dense -> softmax
    kan2_dense_softmax<<<B_TOTAL, 128, 0, stream>>>(o1, kan2_w, dense_w, dense_b, outp);
}
